// DiTBlock_36893769072666
// MI455X (gfx1250) — matrix operand order repaired and checked by reading, not compiled, not run on hardware
//
#include <hip/hip_runtime.h>
#include <hip/hip_bf16.h>
#include <math.h>

// ---------------- constants for this problem instance ----------------
#define B_   2
#define T_   2048
#define D_   1024
#define NH_  16
#define HD_  64
#define HID_ 4096
#define CD_  1024
#define M_   (B_*T_)      // 4096 rows of activations
#define ADA_ (6*D_)       // 6144

typedef __attribute__((ext_vector_type(16))) __bf16 v16bf;
typedef __attribute__((ext_vector_type(8)))  float  v8f;

union FragU { v16bf v; uint4 q[2]; };

// ---------------- bf16 helpers (storage = unsigned short) ----------------
__device__ __forceinline__ unsigned short f2bf(float f) {
    unsigned int u = __float_as_uint(f);
    unsigned int r = u + 0x7fffu + ((u >> 16) & 1u);   // round-to-nearest-even
    return (unsigned short)(r >> 16);
}
__device__ __forceinline__ float bf2f(unsigned short h) {
    return __uint_as_float(((unsigned int)h) << 16);
}

// ---------------- WMMA fragment loaders (ISA 7.12.2 layouts) ----------------
// A (16x32 bf16): lane l -> row m=l&15, h=l>>4; VGPR v0..3: k=8h+2v,+1 ; v4..7: k=16+8h+2v,+1
__device__ __forceinline__ v16bf load_a_frag(const unsigned short* base, int ld, int lane) {
    int m = lane & 15, h = lane >> 4;
    const unsigned short* p = base + (size_t)m * ld + h * 8;
    FragU u;
    u.q[0] = *(const uint4*)(p);
    u.q[1] = *(const uint4*)(p + 16);
    return u.v;
}
// B (32x16 bf16) as W[n][k] (i.e. B = W^T): lane l -> col n=l&15, h=l>>4; VGPR v0..3: k=8h+2v,+1 ; v4..7: k=16+8h+2v,+1
__device__ __forceinline__ v16bf load_b_frag(const unsigned short* base, int ld, int lane) {
    int n = lane & 15, h = lane >> 4;
    const unsigned short* p = base + (size_t)n * ld + h * 8;
    FragU u;
    u.q[0] = *(const uint4*)(p);
    u.q[1] = *(const uint4*)(p + 16);
    return u.v;
}
__device__ __forceinline__ v8f wmma_bf16(v16bf a, v16bf b, v8f c) {
    return __builtin_amdgcn_wmma_f32_16x16x32_bf16(false, a, false, b, (short)0, c, false, false);
}

// ---------------- fp32 -> bf16 weight conversion ----------------
__global__ void cvt_kernel(const float* __restrict__ src, unsigned short* __restrict__ dst, size_t n) {
    size_t i = (size_t)blockIdx.x * blockDim.x + threadIdx.x;
    if (i < n) dst[i] = f2bf(src[i]);
}

// ---------------- ada = silu(cond) @ ada_w^T + ada_b ----------------
__global__ __launch_bounds__(128) void ada_kernel(const float* __restrict__ cond,
                                                  const float* __restrict__ aw,
                                                  const float* __restrict__ ab,
                                                  float* __restrict__ ada) {
    int o = blockIdx.x, b = blockIdx.y;
    const float* c = cond + (size_t)b * CD_;
    const float* w = aw + (size_t)o * CD_;
    float s = 0.f;
    for (int k = threadIdx.x; k < CD_; k += 128) {
        float v = c[k];
        s += (v / (1.f + __expf(-v))) * w[k];
    }
    __shared__ float red[128];
    red[threadIdx.x] = s; __syncthreads();
    for (int st = 64; st; st >>= 1) {
        if (threadIdx.x < st) red[threadIdx.x] += red[threadIdx.x + st];
        __syncthreads();
    }
    if (threadIdx.x == 0) ada[(size_t)b * ADA_ + o] = red[0] + ab[o];
}

// ---------------- h = modulate(rms_norm(x, nw), ada[shift], ada[scale]) -> bf16 ----------------
__global__ __launch_bounds__(256) void rmsmod_kernel(const float* __restrict__ x,
                                                     const float* __restrict__ nw,
                                                     const float* __restrict__ ada,
                                                     int shiftOff, int scaleOff,
                                                     unsigned short* __restrict__ out) {
    int r = blockIdx.x;             // row in [0, M_)
    int b = r / T_;
    const float* row = x + (size_t)r * D_;
    __shared__ float red[256];
    float ss = 0.f;
    for (int c = threadIdx.x; c < D_; c += 256) { float v = row[c]; ss += v * v; }
    red[threadIdx.x] = ss; __syncthreads();
    for (int st = 128; st; st >>= 1) {
        if (threadIdx.x < st) red[threadIdx.x] += red[threadIdx.x + st];
        __syncthreads();
    }
    float rms = rsqrtf(red[0] / (float)D_ + 1e-6f);
    const float* sh = ada + (size_t)b * ADA_ + shiftOff;
    const float* sc = ada + (size_t)b * ADA_ + scaleOff;
    for (int c = threadIdx.x; c < D_; c += 256) {
        float v = row[c] * rms * nw[c];
        v = v * (1.f + sc[c]) + sh[c];
        out[(size_t)r * D_ + c] = f2bf(v);
    }
}

// ---------------- generic bf16 GEMM: C[M,N] = A[M,K] @ W[N,K]^T (+bias) ----------------
// block = 128 threads = 4 waves; wave tile 32x64 (8 WMMAs / K-step); block tile 64x128
__global__ __launch_bounds__(128) void gemm_bf16_kernel(const unsigned short* __restrict__ A, int ldA,
                                                        const unsigned short* __restrict__ W, int ldW,
                                                        const float* __restrict__ bias,
                                                        float* __restrict__ Cf,
                                                        unsigned short* __restrict__ Cb,
                                                        int ldC, int K) {
    int lane = threadIdx.x & 31;
    int wave = threadIdx.x >> 5;
    int m0 = blockIdx.x * 64  + (wave >> 1) * 32;
    int n0 = blockIdx.y * 128 + (wave & 1) * 64;

    v8f acc[2][4] = {};
    const unsigned short* Ar0 = A + (size_t)m0 * ldA;
    const unsigned short* Ar1 = A + (size_t)(m0 + 16) * ldA;
    const unsigned short* Wr0 = W + (size_t)(n0     ) * ldW;
    const unsigned short* Wr1 = W + (size_t)(n0 + 16) * ldW;
    const unsigned short* Wr2 = W + (size_t)(n0 + 32) * ldW;
    const unsigned short* Wr3 = W + (size_t)(n0 + 48) * ldW;

    for (int k = 0; k < K; k += 32) {
        v16bf a0 = load_a_frag(Ar0 + k, ldA, lane);
        v16bf a1 = load_a_frag(Ar1 + k, ldA, lane);
        v16bf b0 = load_b_frag(Wr0 + k, ldW, lane);
        v16bf b1 = load_b_frag(Wr1 + k, ldW, lane);
        v16bf b2 = load_b_frag(Wr2 + k, ldW, lane);
        v16bf b3 = load_b_frag(Wr3 + k, ldW, lane);
        acc[0][0] = wmma_bf16(a0, b0, acc[0][0]);
        acc[0][1] = wmma_bf16(a0, b1, acc[0][1]);
        acc[0][2] = wmma_bf16(a0, b2, acc[0][2]);
        acc[0][3] = wmma_bf16(a0, b3, acc[0][3]);
        acc[1][0] = wmma_bf16(a1, b0, acc[1][0]);
        acc[1][1] = wmma_bf16(a1, b1, acc[1][1]);
        acc[1][2] = wmma_bf16(a1, b2, acc[1][2]);
        acc[1][3] = wmma_bf16(a1, b3, acc[1][3]);
    }

    int nlo = lane & 15, hh = lane >> 4;
    #pragma unroll
    for (int i = 0; i < 2; i++)
        #pragma unroll
        for (int j = 0; j < 4; j++) {
            int n = n0 + j * 16 + nlo;
            float bv = bias ? bias[n] : 0.f;
            #pragma unroll
            for (int v = 0; v < 8; v++) {
                int m = m0 + i * 16 + v + 8 * hh;
                float val = acc[i][j][v] + bv;
                if (Cb) Cb[(size_t)m * ldC + n] = f2bf(val);
                else    Cf[(size_t)m * ldC + n] = val;
            }
        }
}

// ---------------- qk rms_norm + 2-axis RoPE; writes Q[b,h,t,d], K[b,h,t,d], V^T[b,h,d,t] bf16 ----------------
__global__ __launch_bounds__(256) void rope_kernel(const float* __restrict__ qkv,
                                                   const float* __restrict__ qnw,
                                                   const float* __restrict__ knw,
                                                   unsigned short* __restrict__ Qb,
                                                   unsigned short* __restrict__ Kb,
                                                   unsigned short* __restrict__ Vt,
                                                   const int* __restrict__ widthPtr) {
    int wave = threadIdx.x >> 5, lane = threadIdx.x & 31;
    size_t idx = (size_t)blockIdx.x * 8 + wave;     // over B*NH*T
    int t = (int)(idx % T_);
    int h = (int)((idx / T_) % NH_);
    int b = (int)(idx / ((size_t)T_ * NH_));
    int width = *widthPtr;

    // qkv layout: [b, t, s, h, d]
    const float* rq = qkv + ((size_t)(b * T_ + t) * 3 * NH_ + 0 * NH_ + h) * HD_;
    const float* rk = qkv + ((size_t)(b * T_ + t) * 3 * NH_ + 1 * NH_ + h) * HD_;
    const float* rv = qkv + ((size_t)(b * T_ + t) * 3 * NH_ + 2 * NH_ + h) * HD_;

    int d0 = lane * 2;
    float q0 = rq[d0], q1 = rq[d0 + 1];
    float k0 = rk[d0], k1 = rk[d0 + 1];
    float v0 = rv[d0], v1 = rv[d0 + 1];

    float ssq = q0 * q0 + q1 * q1;
    float ssk = k0 * k0 + k1 * k1;
    #pragma unroll
    for (int off = 16; off; off >>= 1) {
        ssq += __shfl_xor(ssq, off);
        ssk += __shfl_xor(ssk, off);
    }
    float rqn = rsqrtf(ssq / (float)HD_ + 1e-6f);
    float rkn = rsqrtf(ssk / (float)HD_ + 1e-6f);
    q0 *= rqn * qnw[d0]; q1 *= rqn * qnw[d0 + 1];
    k0 *= rkn * knw[d0]; k1 *= rkn * knw[d0 + 1];

    // rope: pair index = lane; lane<16 -> y axis, lane>=16 -> x axis; freq index = lane&15 over axis_dim=16
    int fi = lane & 15;
    float coord = (lane < 16) ? (float)(t / width) : (float)(t % width);
    float invf = __expf(-9.210340371976184f * (float)fi / 16.f);  // 10000^(-fi/16)
    float ang = coord * invf;
    float cs = cosf(ang), sn = sinf(ang);
    float oq0 = q0 * cs - q1 * sn, oq1 = q1 * cs + q0 * sn;
    float ok0 = k0 * cs - k1 * sn, ok1 = k1 * cs + k0 * sn;

    size_t qo = (((size_t)b * NH_ + h) * T_ + t) * HD_ + d0;
    Qb[qo]     = f2bf(oq0); Qb[qo + 1] = f2bf(oq1);
    Kb[qo]     = f2bf(ok0); Kb[qo + 1] = f2bf(ok1);
    size_t vo = (((size_t)b * NH_ + h) * HD_ + d0) * T_ + t;
    Vt[vo]      = f2bf(v0);
    Vt[vo + T_] = f2bf(v1);
}

// ---------------- flash attention: one wave per 16-query tile ----------------
__global__ __launch_bounds__(32) void attn_kernel(const unsigned short* __restrict__ Qb,
                                                  const unsigned short* __restrict__ Kb,
                                                  const unsigned short* __restrict__ Vt,
                                                  unsigned short* __restrict__ attnA) {
    __shared__ __align__(16) float Sbuf[16 * 32];
    __shared__ __align__(16) unsigned short Pbuf[16 * 32];
    __shared__ float mrun[16], lrun[16], alphab[16];

    const float scale = 0.125f;   // HD^-0.5
    int lane = threadIdx.x;
    int bh = blockIdx.y;
    int tt = blockIdx.x * 16;

    const unsigned short* Qp = Qb + (size_t)bh * T_ * HD_;
    const unsigned short* Kp = Kb + (size_t)bh * T_ * HD_;
    const unsigned short* Vp = Vt + (size_t)bh * HD_ * T_;

    v16bf aq0 = load_a_frag(Qp + (size_t)tt * HD_ + 0,  HD_, lane);
    v16bf aq1 = load_a_frag(Qp + (size_t)tt * HD_ + 32, HD_, lane);

    if (lane < 16) { mrun[lane] = -__builtin_inff(); lrun[lane] = 0.f; }
    v8f o0 = {}, o1 = {}, o2 = {}, o3 = {};
    __syncthreads();

    int nlo = lane & 15, hh = lane >> 4;
    for (int kb = 0; kb < T_; kb += 32) {
        // S = scale * Q @ K^T for 32 keys (two 16-key subtiles)
        v8f s0 = {}, s1 = {};
        s0 = wmma_bf16(aq0, load_b_frag(Kp + (size_t)kb * HD_ + 0,  HD_, lane), s0);
        s0 = wmma_bf16(aq1, load_b_frag(Kp + (size_t)kb * HD_ + 32, HD_, lane), s0);
        s1 = wmma_bf16(aq0, load_b_frag(Kp + (size_t)(kb + 16) * HD_ + 0,  HD_, lane), s1);
        s1 = wmma_bf16(aq1, load_b_frag(Kp + (size_t)(kb + 16) * HD_ + 32, HD_, lane), s1);

        #pragma unroll
        for (int v = 0; v < 8; v++) {
            int m = v + 8 * hh;
            Sbuf[m * 32 + nlo]      = s0[v] * scale;
            Sbuf[m * 32 + 16 + nlo] = s1[v] * scale;
        }
        __syncthreads();

        // online softmax: lane -> (row m = lane&15, col-half c = lane>>4)
        {
            int m = lane & 15, c = lane >> 4;
            float lmax = -__builtin_inff();
            #pragma unroll
            for (int j = 0; j < 16; j++) lmax = fmaxf(lmax, Sbuf[m * 32 + c * 16 + j]);
            float rowmax = fmaxf(lmax, __shfl_xor(lmax, 16));
            float mnew = fmaxf(mrun[m], rowmax);
            float lsum = 0.f;
            #pragma unroll
            for (int j = 0; j < 16; j++) {
                float p = __expf(Sbuf[m * 32 + c * 16 + j] - mnew);
                lsum += p;
                Pbuf[m * 32 + c * 16 + j] = f2bf(p);
            }
            lsum += __shfl_xor(lsum, 16);
            float alpha = __expf(mrun[m] - mnew);
            if (c == 0) {
                alphab[m] = alpha;
                lrun[m] = lrun[m] * alpha + lsum;
                mrun[m] = mnew;
            }
        }
        __syncthreads();

        // rescale running O
        #pragma unroll
        for (int v = 0; v < 8; v++) {
            float f = alphab[v + 8 * hh];
            o0[v] *= f; o1[v] *= f; o2[v] *= f; o3[v] *= f;
        }

        // P as A-fragment from LDS
        v16bf pf;
        {
            int m = lane & 15, h = lane >> 4;
            FragU u;
            u.q[0] = *(const uint4*)&Pbuf[m * 32 + h * 8];
            u.q[1] = *(const uint4*)&Pbuf[m * 32 + 16 + h * 8];
            pf = u.v;
        }
        // O += P @ V  (V^T rows are head dims -> contiguous B-frag loads)
        o0 = wmma_bf16(pf, load_b_frag(Vp + (size_t)0  * T_ + kb, T_, lane), o0);
        o1 = wmma_bf16(pf, load_b_frag(Vp + (size_t)16 * T_ + kb, T_, lane), o1);
        o2 = wmma_bf16(pf, load_b_frag(Vp + (size_t)32 * T_ + kb, T_, lane), o2);
        o3 = wmma_bf16(pf, load_b_frag(Vp + (size_t)48 * T_ + kb, T_, lane), o3);
        __syncthreads();
    }

    float rl[8];
    #pragma unroll
    for (int v = 0; v < 8; v++) rl[v] = 1.f / lrun[v + 8 * hh];
    int b = bh / NH_, h = bh % NH_;
    v8f oo[4] = {o0, o1, o2, o3};
    #pragma unroll
    for (int j = 0; j < 4; j++)
        #pragma unroll
        for (int v = 0; v < 8; v++) {
            int t = tt + v + 8 * hh;
            attnA[((size_t)b * T_ + t) * D_ + h * HD_ + j * 16 + nlo] = f2bf(oo[j][v] * rl[v]);
        }
}

// ---------------- out = base + ada[gate] * C ----------------
__global__ void resid_kernel(const float* __restrict__ base, const float* __restrict__ C,
                             const float* __restrict__ ada, int gateOff,
                             float* __restrict__ out) {
    size_t i = (size_t)blockIdx.x * blockDim.x + threadIdx.x;
    if (i >= (size_t)M_ * D_) return;
    int c = (int)(i % D_);
    int b = (int)(i / ((size_t)T_ * D_));
    out[i] = base[i] + ada[(size_t)b * ADA_ + gateOff + c] * C[i];
}

// ---------------- in-place SwiGLU on fc1 output (bf16, [M, 2*HID]) ----------------
__global__ void swiglu_kernel(unsigned short* __restrict__ f) {
    size_t i = (size_t)blockIdx.x * blockDim.x + threadIdx.x;
    if (i >= (size_t)M_ * HID_) return;
    size_t r = i / HID_;
    int c = (int)(i % HID_);
    size_t base = r * (size_t)(2 * HID_);
    float v = bf2f(f[base + c]);
    float g = bf2f(f[base + HID_ + c]);
    f[base + c] = f2bf(v * (g / (1.f + __expf(-g))));
}

// ---------------- host orchestration ----------------
extern "C" void kernel_launch(void* const* d_in, const int* in_sizes, int n_in,
                              void* d_out, int out_size, void* d_ws, size_t ws_size,
                              hipStream_t stream) {
    (void)in_sizes; (void)n_in; (void)out_size; (void)ws_size;
    const float* x      = (const float*)d_in[0];
    const float* cond   = (const float*)d_in[1];
    const float* norm1w = (const float*)d_in[2];
    const float* qkvw   = (const float*)d_in[3];
    const float* qnw    = (const float*)d_in[4];
    const float* knw    = (const float*)d_in[5];
    const float* projw  = (const float*)d_in[6];
    const float* projb  = (const float*)d_in[7];
    const float* norm2w = (const float*)d_in[8];
    const float* fc1w   = (const float*)d_in[9];
    const float* fc1b   = (const float*)d_in[10];
    const float* fc2w   = (const float*)d_in[11];
    const float* fc2b   = (const float*)d_in[12];
    const float* adaw   = (const float*)d_in[13];
    const float* adab   = (const float*)d_in[14];
    const int*   widthP = (const int*)d_in[16];
    float* out = (float*)d_out;
    char* ws = (char*)d_ws;

    size_t off = 0;
    auto alloc = [&](size_t bytes) { size_t o = off; off += (bytes + 255) & ~(size_t)255; return o; };

    size_t o_ada    = alloc((size_t)B_ * ADA_ * 4);
    size_t o_qkvwb  = alloc((size_t)3 * D_ * D_ * 2);
    size_t o_projwb = alloc((size_t)D_ * D_ * 2);
    size_t o_fc1wb  = alloc((size_t)2 * HID_ * D_ * 2);
    size_t o_fc2wb  = alloc((size_t)D_ * HID_ * 2);
    size_t o_Abuf   = alloc((size_t)M_ * D_ * 2);
    size_t o_Qb     = alloc((size_t)M_ * D_ * 2);
    size_t o_Kb     = alloc((size_t)M_ * D_ * 2);
    size_t o_Vt     = alloc((size_t)M_ * D_ * 2);
    size_t o_attnA  = alloc((size_t)M_ * D_ * 2);
    size_t o_x2     = alloc((size_t)M_ * D_ * 4);
    size_t o_gemmC  = alloc((size_t)M_ * D_ * 4);
    size_t o_slab   = alloc((size_t)M_ * 2 * HID_ * 2);  // qkvC fp32 (48MB) early, fc1out bf16 (64MB) later

    float* ada            = (float*)(ws + o_ada);
    unsigned short* qkvwb = (unsigned short*)(ws + o_qkvwb);
    unsigned short* prjwb = (unsigned short*)(ws + o_projwb);
    unsigned short* fc1wb = (unsigned short*)(ws + o_fc1wb);
    unsigned short* fc2wb = (unsigned short*)(ws + o_fc2wb);
    unsigned short* Abuf  = (unsigned short*)(ws + o_Abuf);
    unsigned short* Qb    = (unsigned short*)(ws + o_Qb);
    unsigned short* Kb    = (unsigned short*)(ws + o_Kb);
    unsigned short* Vt    = (unsigned short*)(ws + o_Vt);
    unsigned short* attnA = (unsigned short*)(ws + o_attnA);
    float* x2             = (float*)(ws + o_x2);
    float* gemmC          = (float*)(ws + o_gemmC);
    float* qkvC           = (float*)(ws + o_slab);
    unsigned short* fc1o  = (unsigned short*)(ws + o_slab);

    auto blks = [](size_t n, int bs) { return (unsigned)((n + bs - 1) / bs); };

    // weights -> bf16
    cvt_kernel<<<blks((size_t)3*D_*D_, 256), 256, 0, stream>>>(qkvw, qkvwb, (size_t)3*D_*D_);
    cvt_kernel<<<blks((size_t)D_*D_, 256),   256, 0, stream>>>(projw, prjwb, (size_t)D_*D_);
    cvt_kernel<<<blks((size_t)2*HID_*D_,256),256, 0, stream>>>(fc1w, fc1wb, (size_t)2*HID_*D_);
    cvt_kernel<<<blks((size_t)D_*HID_, 256), 256, 0, stream>>>(fc2w, fc2wb, (size_t)D_*HID_);

    // adaLN
    ada_kernel<<<dim3(ADA_, B_), 128, 0, stream>>>(cond, adaw, adab, ada);

    // norm1 + modulate -> bf16 A
    rmsmod_kernel<<<M_, 256, 0, stream>>>(x, norm1w, ada, 0, D_, Abuf);

    // qkv = A @ qkv_w^T   (4096 x 3072 x 1024)
    gemm_bf16_kernel<<<dim3(M_/64, 3*D_/128), 128, 0, stream>>>(Abuf, D_, qkvwb, D_,
                                                                nullptr, qkvC, nullptr, 3*D_, D_);
    // qk-norm + rope + V transpose
    rope_kernel<<<(B_*NH_*T_)/8, 256, 0, stream>>>(qkvC, qnw, knw, Qb, Kb, Vt, widthP);

    // attention
    attn_kernel<<<dim3(T_/16, B_*NH_), 32, 0, stream>>>(Qb, Kb, Vt, attnA);

    // proj (+bias)
    gemm_bf16_kernel<<<dim3(M_/64, D_/128), 128, 0, stream>>>(attnA, D_, prjwb, D_,
                                                              projb, gemmC, nullptr, D_, D_);
    // x2 = x + g_a * proj
    resid_kernel<<<blks((size_t)M_*D_, 256), 256, 0, stream>>>(x, gemmC, ada, 2*D_, x2);

    // norm2 + modulate -> bf16 A
    rmsmod_kernel<<<M_, 256, 0, stream>>>(x2, norm2w, ada, 3*D_, 4*D_, Abuf);

    // fc1 (+bias) -> bf16 [M, 2*HID]
    gemm_bf16_kernel<<<dim3(M_/64, 2*HID_/128), 128, 0, stream>>>(Abuf, D_, fc1wb, D_,
                                                                  fc1b, nullptr, fc1o, 2*HID_, D_);
    // swiglu in place (value half)
    swiglu_kernel<<<blks((size_t)M_*HID_, 256), 256, 0, stream>>>(fc1o);

    // fc2 (+bias): A = fc1o with ldA = 2*HID
    gemm_bf16_kernel<<<dim3(M_/64, D_/128), 128, 0, stream>>>(fc1o, 2*HID_, fc2wb, HID_,
                                                              fc2b, gemmC, nullptr, D_, HID_);
    // out = x2 + g_m * fc2
    resid_kernel<<<blks((size_t)M_*D_, 256), 256, 0, stream>>>(x2, gemmC, ada, 5*D_, out);
}
